// LigerLMHeadCE_49795850830127
// MI455X (gfx1250) — matrix operand order repaired and checked by reading, not compiled, not run on hardware
//
#include <hip/hip_runtime.h>
#include <math.h>

#define IGNORE_INDEX (-100)

typedef __bf16 bf16_t;
typedef __attribute__((ext_vector_type(16))) __bf16 v16bf;
typedef __attribute__((ext_vector_type(8)))  __bf16 v8bf;
typedef __attribute__((ext_vector_type(4)))  __bf16 v4bf;
typedef __attribute__((ext_vector_type(8)))  float  v8f;
typedef __attribute__((ext_vector_type(4)))  float  v4f;

static constexpr int N_TOK = 2048;
static constexpr int H_DIM = 2048;
static constexpr int V_DIM = 50257;
static constexpr int BM = 128;   // token rows per block
static constexpr int BV = 128;   // vocab cols per block
static constexpr int BK = 32;    // K tile (one wmma K step)
static constexpr int LDT = BK + 8;                  // padded LDS row (bf16 elems)
static constexpr int NVB = (V_DIM + BV - 1) / BV;   // 393 vocab blocks

// ---------------------------------------------------------------------------
// Kernel 1: tiled bf16x3 WMMA GEMM fused with per-row block max / sum(exp).
// Software-pipelined: next K-tile is prefetched into registers while the
// current tile's WMMAs run, hiding L2/HBM latency behind matrix compute.
// ---------------------------------------------------------------------------
__global__ __launch_bounds__(256)
void lmhead_partial_kernel(const float* __restrict__ x,
                           const float* __restrict__ w,
                           float* __restrict__ pmax,
                           float* __restrict__ psum)
{
    __shared__ __align__(16) bf16_t sAh[BM * LDT];
    __shared__ __align__(16) bf16_t sAl[BM * LDT];
    __shared__ __align__(16) bf16_t sBh[BV * LDT];
    __shared__ __align__(16) bf16_t sBl[BV * LDT];
    __shared__ float sred[2][2][BM];   // [max|sum][wv][row]

    const int tid  = threadIdx.x;
    const int lane = tid & 31;
    const int wave = tid >> 5;
    const int wm   = wave & 3;    // 4 row-waves  * 32 rows
    const int wv   = wave >> 2;   // 2 col-waves  * 64 cols
    const int hlf  = lane >> 4;   // lane half (WMMA layout)
    const int ln   = lane & 15;

    const int m0 = blockIdx.x * BM;
    const int v0 = blockIdx.y * BV;

    v8f acc[2][4];
#pragma unroll
    for (int mi = 0; mi < 2; ++mi)
#pragma unroll
        for (int vi = 0; vi < 4; ++vi)
#pragma unroll
            for (int e = 0; e < 8; ++e) acc[mi][vi][e] = 0.0f;

    // ---- producer: fixed per-thread addresses (4 float4 each of A and B) ----
    int soff[4];
    const float* pA[4];
    const float* pB[4];
#pragma unroll
    for (int i = 0; i < 4; ++i) {
        const int fid = tid + i * 256;        // 0..1023 float4 slots
        const int row = fid >> 3;             // 8 float4 per 32-float row
        const int col = (fid & 7) << 2;
        soff[i] = row * LDT + col;
        pA[i] = x + (size_t)(m0 + row) * H_DIM + col;
        int vr = v0 + row;                    // clamp ragged vocab tail
        if (vr >= V_DIM) vr = V_DIM - 1;
        pB[i] = w + (size_t)vr * H_DIM + col;
    }

    // prologue: prefetch tile k0 = 0
    v4f ra[4], rb[4];
#pragma unroll
    for (int i = 0; i < 4; ++i) {
        ra[i] = *(const v4f*)(pA[i]);
        rb[i] = *(const v4f*)(pB[i]);
    }

    for (int k0 = 0; k0 < H_DIM; k0 += BK) {
        // ---- convert prefetched fp32 -> hi/lo bf16, store to LDS ----
#pragma unroll
        for (int i = 0; i < 4; ++i) {
            v4bf h, l;
#pragma unroll
            for (int e = 0; e < 4; ++e) {
                const float f   = ra[i][e];
                const bf16_t hb = (bf16_t)f;
                h[e] = hb;
                l[e] = (bf16_t)(f - (float)hb);
            }
            *(v4bf*)&sAh[soff[i]] = h;
            *(v4bf*)&sAl[soff[i]] = l;
#pragma unroll
            for (int e = 0; e < 4; ++e) {
                const float f   = rb[i][e];
                const bf16_t hb = (bf16_t)f;
                h[e] = hb;
                l[e] = (bf16_t)(f - (float)hb);
            }
            *(v4bf*)&sBh[soff[i]] = h;
            *(v4bf*)&sBl[soff[i]] = l;
        }
        __syncthreads();

        // ---- prefetch next K-tile; latency overlaps the WMMA phase below ----
        const int kn = k0 + BK;
        if (kn < H_DIM) {
#pragma unroll
            for (int i = 0; i < 4; ++i) {
                ra[i] = *(const v4f*)(pA[i] + kn);
                rb[i] = *(const v4f*)(pB[i] + kn);
            }
        }

        // ---- A fragments (16x32 bf16: K = 8h..8h+7 and 16+8h..23+8h) ----
        v16bf ah[2], al[2];
#pragma unroll
        for (int mi = 0; mi < 2; ++mi) {
            const int r = (wm * 32 + mi * 16 + ln) * LDT;
            v8bf h0 = *(const v8bf*)&sAh[r + 8 * hlf];
            v8bf h1 = *(const v8bf*)&sAh[r + 16 + 8 * hlf];
            ah[mi] = __builtin_shufflevector(h0, h1, 0,1,2,3,4,5,6,7,8,9,10,11,12,13,14,15);
            v8bf l0 = *(const v8bf*)&sAl[r + 8 * hlf];
            v8bf l1 = *(const v8bf*)&sAl[r + 16 + 8 * hlf];
            al[mi] = __builtin_shufflevector(l0, l1, 0,1,2,3,4,5,6,7,8,9,10,11,12,13,14,15);
        }

        // ---- B fragments (32x16: K = 8h..8h+7 and 16+8h..23+8h) + WMMA ----
#pragma unroll
        for (int vi = 0; vi < 4; ++vi) {
            const int r = (wv * 64 + vi * 16 + ln) * LDT;
            v8bf h0 = *(const v8bf*)&sBh[r + 8 * hlf];
            v8bf h1 = *(const v8bf*)&sBh[r + 16 + 8 * hlf];
            v16bf bh = __builtin_shufflevector(h0, h1, 0,1,2,3,4,5,6,7,8,9,10,11,12,13,14,15);
            v8bf l0 = *(const v8bf*)&sBl[r + 8 * hlf];
            v8bf l1 = *(const v8bf*)&sBl[r + 16 + 8 * hlf];
            v16bf bl = __builtin_shufflevector(l0, l1, 0,1,2,3,4,5,6,7,8,9,10,11,12,13,14,15);
#pragma unroll
            for (int mi = 0; mi < 2; ++mi) {
                acc[mi][vi] = __builtin_amdgcn_wmma_f32_16x16x32_bf16(
                    false, ah[mi], false, bh, (short)0, acc[mi][vi], false, false);
                acc[mi][vi] = __builtin_amdgcn_wmma_f32_16x16x32_bf16(
                    false, ah[mi], false, bl, (short)0, acc[mi][vi], false, false);
                acc[mi][vi] = __builtin_amdgcn_wmma_f32_16x16x32_bf16(
                    false, al[mi], false, bh, (short)0, acc[mi][vi], false, false);
            }
        }
        __syncthreads();
    }

    // ---- fused per-row block reduction: max, then sum(exp(.-max)) ----
    // C layout: lane (hlf,ln) VGPR j -> row = j + 8*hlf, col = ln (within 16x16)
    float gm[2][8];
#pragma unroll
    for (int mi = 0; mi < 2; ++mi)
#pragma unroll
        for (int j = 0; j < 8; ++j) {
            float m = -INFINITY;
#pragma unroll
            for (int vi = 0; vi < 4; ++vi) {
                const int cg = v0 + wv * 64 + vi * 16 + ln;
                const float v = acc[mi][vi][j];
                if (cg < V_DIM && v > m) m = v;
            }
#pragma unroll
            for (int s = 1; s < 16; s <<= 1) {
                const float o = __shfl_xor(m, s, 32);
                if (o > m) m = o;
            }
            gm[mi][j] = m;   // wave-strip max (64 cols)
        }
    if (ln == 0) {
#pragma unroll
        for (int mi = 0; mi < 2; ++mi)
#pragma unroll
            for (int j = 0; j < 8; ++j)
                sred[0][wv][wm * 32 + mi * 16 + j + 8 * hlf] = gm[mi][j];
    }
    __syncthreads();

    float sm[2][8];
#pragma unroll
    for (int mi = 0; mi < 2; ++mi)
#pragma unroll
        for (int j = 0; j < 8; ++j) {
            const int row = wm * 32 + mi * 16 + j + 8 * hlf;
            const float M = fmaxf(sred[0][0][row], sred[0][1][row]);
            gm[mi][j] = M;   // block-wide row max
            float s = 0.0f;
#pragma unroll
            for (int vi = 0; vi < 4; ++vi) {
                const int cg = v0 + wv * 64 + vi * 16 + ln;
                if (cg < V_DIM) s += expf(acc[mi][vi][j] - M);
            }
#pragma unroll
            for (int t = 1; t < 16; t <<= 1) s += __shfl_xor(s, t, 32);
            sm[mi][j] = s;
        }
    if (ln == 0) {
#pragma unroll
        for (int mi = 0; mi < 2; ++mi)
#pragma unroll
            for (int j = 0; j < 8; ++j)
                sred[1][wv][wm * 32 + mi * 16 + j + 8 * hlf] = sm[mi][j];
    }
    __syncthreads();

    if (wv == 0 && ln == 0) {
#pragma unroll
        for (int mi = 0; mi < 2; ++mi)
#pragma unroll
            for (int j = 0; j < 8; ++j) {
                const int row = wm * 32 + mi * 16 + j + 8 * hlf;
                // [vblock][row] layout -> coalesced reads in finalize
                const size_t idx = (size_t)blockIdx.y * N_TOK + (m0 + row);
                pmax[idx] = gm[mi][j];
                psum[idx] = sred[1][0][row] + sred[1][1][row];
            }
    }
}

// ---------------------------------------------------------------------------
// Kernel 2: exact fp32 gold logit  gold[n] = dot(x[n], W[y[n]])
// ---------------------------------------------------------------------------
__global__ __launch_bounds__(256)
void gold_kernel(const float* __restrict__ x, const float* __restrict__ w,
                 const int* __restrict__ y, float* __restrict__ gold)
{
    const int n  = blockIdx.x;
    const int yn = y[n];
    const int t  = threadIdx.x;
    float s = 0.0f;
    if (yn != IGNORE_INDEX) {
        const float* xr = x + (size_t)n * H_DIM;
        const float* wr = w + (size_t)yn * H_DIM;
        for (int k = t * 4; k < H_DIM; k += 256 * 4) {
            const v4f a = *(const v4f*)(xr + k);
            const v4f b = *(const v4f*)(wr + k);
            s += a[0] * b[0] + a[1] * b[1] + a[2] * b[2] + a[3] * b[3];
        }
    }
    for (int m = 1; m < 32; m <<= 1) s += __shfl_xor(s, m, 32);
    __shared__ float sb[8];
    if ((t & 31) == 0) sb[t >> 5] = s;
    __syncthreads();
    if (t == 0) {
        float tot = 0.0f;
        for (int i = 0; i < 8; ++i) tot += sb[i];
        gold[n] = tot;
    }
}

// ---------------------------------------------------------------------------
// Kernel 3a: per-row lse + masked NLL, 8 blocks, coalesced [vblock][row] reads,
// deterministic per-block partial sums (no float atomics).
// ---------------------------------------------------------------------------
__global__ __launch_bounds__(256)
void finalize_rows_kernel(const int* __restrict__ y,
                          const float* __restrict__ pmax,
                          const float* __restrict__ psum,
                          const float* __restrict__ gold,
                          float* __restrict__ part_sum,
                          float* __restrict__ part_cnt)
{
    const int n = blockIdx.x * 256 + threadIdx.x;   // N_TOK == 8*256
    float nll = 0.0f;
    int   cnt = 0;
    {
        float M = -INFINITY;
        for (int b = 0; b < NVB; ++b)
            M = fmaxf(M, pmax[(size_t)b * N_TOK + n]);
        float s = 0.0f;
        for (int b = 0; b < NVB; ++b)
            s += psum[(size_t)b * N_TOK + n] * expf(pmax[(size_t)b * N_TOK + n] - M);
        const float lse = M + logf(s);
        if (y[n] != IGNORE_INDEX) { nll = lse - gold[n]; cnt = 1; }
    }
    for (int m = 1; m < 32; m <<= 1) {
        nll += __shfl_xor(nll, m, 32);
        cnt += __shfl_xor(cnt, m, 32);
    }
    __shared__ float fs[8];
    __shared__ int   fc[8];
    if ((threadIdx.x & 31) == 0) { fs[threadIdx.x >> 5] = nll; fc[threadIdx.x >> 5] = cnt; }
    __syncthreads();
    if (threadIdx.x == 0) {
        float S = 0.0f; int C = 0;
        for (int i = 0; i < 8; ++i) { S += fs[i]; C += fc[i]; }
        part_sum[blockIdx.x] = S;
        part_cnt[blockIdx.x] = (float)C;
    }
}

// ---------------------------------------------------------------------------
// Kernel 3b: combine 8 partials -> mean NLL
// ---------------------------------------------------------------------------
__global__ void finalize_out_kernel(const float* __restrict__ part_sum,
                                    const float* __restrict__ part_cnt,
                                    float* __restrict__ out)
{
    if (threadIdx.x == 0) {
        float S = 0.0f, C = 0.0f;
        for (int i = 0; i < 8; ++i) { S += part_sum[i]; C += part_cnt[i]; }
        out[0] = S / fmaxf(C, 1.0f);
    }
}

// ---------------------------------------------------------------------------
extern "C" void kernel_launch(void* const* d_in, const int* in_sizes, int n_in,
                              void* d_out, int out_size, void* d_ws, size_t ws_size,
                              hipStream_t stream)
{
    (void)in_sizes; (void)n_in; (void)out_size; (void)ws_size;
    const float* x = (const float*)d_in[0];
    const float* w = (const float*)d_in[1];
    const int*   y = (const int*)d_in[2];
    float* out = (float*)d_out;

    float* ws    = (float*)d_ws;
    float* pmax  = ws;                                   // NVB * N_TOK
    float* psum  = ws + (size_t)NVB * N_TOK;             // NVB * N_TOK
    float* gold  = ws + (size_t)2 * NVB * N_TOK;         // N_TOK
    float* psums = gold + N_TOK;                         // 8
    float* pcnts = psums + 8;                            // 8
    // total workspace: (2*393*2048 + 2048 + 16)*4 B ~= 6.45 MB

    dim3 g1(N_TOK / BM, NVB);   // m-block fast => weight strip reused via L2
    lmhead_partial_kernel<<<g1, 256, 0, stream>>>(x, w, pmax, psum);
    gold_kernel<<<N_TOK, 256, 0, stream>>>(x, w, y, gold);
    finalize_rows_kernel<<<N_TOK / 256, 256, 0, stream>>>(y, pmax, psum, gold, psums, pcnts);
    finalize_out_kernel<<<1, 32, 0, stream>>>(psums, pcnts, out);
}
